// ThreeBodyLayer_66537633349676
// MI455X (gfx1250) — matrix operand order repaired and checked by reading, not compiled, not run on hardware
//
#include <hip/hip_runtime.h>

#define HID   48
#define XK    64      // GEMM1 K padded 48 -> 64 (2 k-steps of 32)
#define CK    96      // GEMM2 K = 96 (3 k-steps of 32)
#define TILE  16      // triples per wave-tile
#define WAVES 4       // waves per block (128 threads)

typedef __attribute__((ext_vector_type(16))) _Float16 v16h;
typedef __attribute__((ext_vector_type(8)))  float    v8f;
typedef __attribute__((ext_vector_type(2)))  __fp16   fp16x2;

union Frag16 { v16h v; uint4 q[2]; };

// Pack 4 f32 -> 4 f16 (2x v_cvt_pkrtz) as uint2 for an 8-byte LDS store.
__device__ inline uint2 pk4(float a, float b, float c, float d) {
  union { fp16x2 v; unsigned u; } lo, hi;
  lo.v = __builtin_amdgcn_cvt_pkrtz(a, b);
  hi.v = __builtin_amdgcn_cvt_pkrtz(c, d);
  uint2 r; r.x = lo.u; r.y = hi.u; return r;
}

// A-matrix fragment (16x32 f16) from row-major f16 tile in LDS (contiguous k order, same as load_bfrag):
// lanes 0-15: row = lane,   K = kb..kb+15
// lanes16-31: row = lane&15, K = kb+16..kb+31
__device__ inline v16h load_afrag(const _Float16* base, int stride, int row,
                                  int kb, int hi) {
  Frag16 f;
  const _Float16* p = base + row * stride + kb + hi * 16;
  f.q[0] = *reinterpret_cast<const uint4*>(p);
  f.q[1] = *reinterpret_cast<const uint4*>(p + 8);
  return f.v;
}

// B-matrix fragment (32x16 f16) for B[k][n] = W[n][k], W row-major f16:
// lane: N = lane&15 (= weight row), K = kb + 16*hi + j -> 16 contiguous halves.
__device__ inline v16h load_bfrag(const _Float16* W, int stride, int nrow,
                                  int kb, int hi) {
  Frag16 f;
  const _Float16* p = W + nrow * stride + kb + hi * 16;
  f.q[0] = *reinterpret_cast<const uint4*>(p);
  f.q[1] = *reinterpret_cast<const uint4*>(p + 8);
  return f.v;
}

// ---------------- workspace zeroing ----------------
__global__ void zero_kernel(float* p, long n) {
  long i = (long)blockIdx.x * blockDim.x + threadIdx.x;
  long stride = (long)gridDim.x * blockDim.x;
  for (; i < n; i += stride) p[i] = 0.0f;
}

// ---------------- weight conversion to f16 ----------------
__global__ void prep_kernel(const float* __restrict__ W3,
                            const float* __restrict__ Wu,
                            _Float16* __restrict__ W3h,
                            _Float16* __restrict__ Wuh) {
  int t = blockIdx.x * blockDim.x + threadIdx.x;
  int stride = gridDim.x * blockDim.x;
  for (int e = t; e < HID * XK; e += stride) {
    int o = e / XK, k = e % XK;
    W3h[e] = (k < HID) ? (_Float16)W3[o * HID + k] : (_Float16)0.0f;
  }
  for (int e = t; e < HID * CK; e += stride)
    Wuh[e] = (_Float16)Wu[e];
}

// ---------------- main triple kernel (WMMA) ----------------
__global__ __launch_bounds__(WAVES * 32) void
triples_kernel(const float* __restrict__ h,
               const int* __restrict__ tb_index,
               const float* __restrict__ d1,
               const float* __restrict__ d2,
               const _Float16* __restrict__ W3h,
               const _Float16* __restrict__ Wuh,
               float* __restrict__ agg,
               float* __restrict__ nc,
               int n_triples, int ntiles, int iters, int wavesTotal) {
  __shared__ _Float16 sX[WAVES][TILE][XK];   // h[nb1]+h[nb2], f16, K-padded
  __shared__ _Float16 sC[WAVES][TILE][CK];   // [h[center] | msg], f16
  __shared__ float    sW[WAVES][TILE];       // per-triple weight
  __shared__ int      sCen[WAVES][TILE], sN1[WAVES][TILE], sN2[WAVES][TILE];

  const int lane  = threadIdx.x & 31;
  const int w     = threadIdx.x >> 5;
  const int gwave = blockIdx.x * WAVES + w;
  const int hi    = lane >> 4;      // half-wave select
  const int ln    = lane & 15;

  // Persistent W3 B-fragments only (48 VGPRs); Wu reloaded per tile (L1-hot).
  v16h b3[3][2];
#pragma unroll
  for (int n = 0; n < 3; ++n)
#pragma unroll
    for (int k = 0; k < 2; ++k)
      b3[n][k] = load_bfrag(W3h, XK, n * 16 + ln, k * 32, hi);

  // X zero-padding (cols 48..63) is loop-invariant: write once.
  {
    uint4 z; z.x = z.y = z.z = z.w = 0u;
    *reinterpret_cast<uint4*>(&sX[w][lane >> 1][HID + (lane & 1) * 8]) = z;
  }

  for (int it = 0; it < iters; ++it) {
    const int tile    = it * wavesTotal + gwave;
    const bool active = tile < ntiles;        // wave-uniform
    const int tbase   = tile * TILE;

    if (active) {
      // --- per-triple metadata (lanes 0..15 own one triple each) ---
      if (lane < TILE) {
        int t = tbase + lane;
        int c = 0, a = 0, b = 0;
        float wt = 0.0f;
        if (t < n_triples) {
          c = tb_index[3 * t + 0];
          a = tb_index[3 * t + 1];
          b = tb_index[3 * t + 2];
          float x1 = d1[t], x2 = d2[t];
          float asym = fabsf(x1 - x2) / (fmaxf(x1, x2) + 1e-8f);
          wt = 1.0f + 0.3f * asym;
          atomicAdd(&nc[c], 1.0f);
        }
        sCen[w][lane] = c; sN1[w][lane] = a; sN2[w][lane] = b;
        sW[w][lane] = wt;
      }
      // --- cooperative gather, float4-vectorized: 16 rows x 12 chunks ---
#pragma unroll
      for (int j = 0; j < (TILE * HID / 4) / 32; ++j) {  // 6 steps
        int e = j * 32 + lane;
        int row = e / (HID / 4), c4 = e % (HID / 4);
        const float4 a = *reinterpret_cast<const float4*>(
            h + (long)sN1[w][row] * HID + c4 * 4);
        const float4 b = *reinterpret_cast<const float4*>(
            h + (long)sN2[w][row] * HID + c4 * 4);
        const float4 c = *reinterpret_cast<const float4*>(
            h + (long)sCen[w][row] * HID + c4 * 4);
        *reinterpret_cast<uint2*>(&sX[w][row][c4 * 4]) =
            pk4(a.x + b.x, a.y + b.y, a.z + b.z, a.w + b.w);
        *reinterpret_cast<uint2*>(&sC[w][row][c4 * 4]) =
            pk4(c.x, c.y, c.z, c.w);
      }
    }

    // --- GEMM1: msg = X @ W3^T (EXEC all-1s, uniform path) ---
    v16h a0 = load_afrag(&sX[w][0][0], XK, ln, 0, hi);
    v16h a1 = load_afrag(&sX[w][0][0], XK, ln, 32, hi);
#pragma unroll
    for (int n = 0; n < 3; ++n) {
      v8f acc = {};
      acc = __builtin_amdgcn_wmma_f32_16x16x32_f16(false, a0, false, b3[n][0],
                                                   (short)0, acc, false, false);
      acc = __builtin_amdgcn_wmma_f32_16x16x32_f16(false, a1, false, b3[n][1],
                                                   (short)0, acc, false, false);
      // D layout: VGPR r -> M = r + 8*hi, N = ln. Spill msg tile as f16 into sC.
#pragma unroll
      for (int r = 0; r < 8; ++r)
        sC[w][r + hi * 8][HID + n * 16 + ln] = (_Float16)acc[r];
    }

    // --- GEMM2: pre = [h_center | msg] @ Wu^T; Wu frags streamed from L1 ---
    v16h c0 = load_afrag(&sC[w][0][0], CK, ln, 0, hi);
    v16h c1 = load_afrag(&sC[w][0][0], CK, ln, 32, hi);
    v16h c2 = load_afrag(&sC[w][0][0], CK, ln, 64, hi);
    v8f acc2[3];
#pragma unroll
    for (int n = 0; n < 3; ++n) {
      v8f acc = {};
      acc = __builtin_amdgcn_wmma_f32_16x16x32_f16(
          false, c0, false, load_bfrag(Wuh, CK, n * 16 + ln, 0, hi),
          (short)0, acc, false, false);
      acc = __builtin_amdgcn_wmma_f32_16x16x32_f16(
          false, c1, false, load_bfrag(Wuh, CK, n * 16 + ln, 32, hi),
          (short)0, acc, false, false);
      acc = __builtin_amdgcn_wmma_f32_16x16x32_f16(
          false, c2, false, load_bfrag(Wuh, CK, n * 16 + ln, 64, hi),
          (short)0, acc, false, false);
      acc2[n] = acc;
    }

    // --- epilogue: leaky-relu * weight, scatter via f32 atomics (L2-resident) ---
    if (active) {
      float wreg[8];
      int   creg[8];
#pragma unroll
      for (int r = 0; r < 8; ++r) {
        wreg[r] = sW[w][r + hi * 8];
        creg[r] = sCen[w][r + hi * 8];
      }
#pragma unroll
      for (int n = 0; n < 3; ++n) {
#pragma unroll
        for (int r = 0; r < 8; ++r) {
          if (tbase + r + hi * 8 < n_triples) {
            float pre = acc2[n][r];
            float dlt = (pre > 0.0f ? pre : 0.01f * pre) * wreg[r];
            atomicAdd(&agg[(long)creg[r] * HID + n * 16 + ln], dlt);
          }
        }
      }
    }
  }
}

// ---------------- finalize: agg/sqrt(nc), residual, layernorm ----------------
__global__ void finalize_kernel(const float* __restrict__ h,
                                const float* __restrict__ agg,
                                const float* __restrict__ nc,
                                const float* __restrict__ gamma,
                                const float* __restrict__ beta,
                                float* __restrict__ out, int n_nodes) {
  int i = blockIdx.x * blockDim.x + threadIdx.x;
  if (i >= n_nodes) return;
  float inv = rsqrtf(fmaxf(nc[i], 1.0f));
  float x[HID];
  float mu = 0.0f;
#pragma unroll
  for (int j = 0; j < HID; ++j) {
    x[j] = h[(long)i * HID + j] + agg[(long)i * HID + j] * inv;
    mu += x[j];
  }
  mu *= (1.0f / HID);
  float var = 0.0f;
#pragma unroll
  for (int j = 0; j < HID; ++j) { float d = x[j] - mu; var += d * d; }
  var *= (1.0f / HID);
  float s = rsqrtf(var + 1e-5f);
#pragma unroll
  for (int j = 0; j < HID; ++j)
    out[(long)i * HID + j] = (x[j] - mu) * s * gamma[j] + beta[j];
}

extern "C" void kernel_launch(void* const* d_in, const int* in_sizes, int n_in,
                              void* d_out, int out_size, void* d_ws, size_t ws_size,
                              hipStream_t stream) {
  const float* h        = (const float*)d_in[0];
  const int*   tb_index = (const int*)d_in[1];
  const float* tb_d1    = (const float*)d_in[2];
  const float* tb_d2    = (const float*)d_in[3];
  const float* W3       = (const float*)d_in[4];
  const float* Wu       = (const float*)d_in[5];
  const float* gamma    = (const float*)d_in[6];
  const float* beta     = (const float*)d_in[7];
  float* out = (float*)d_out;

  const int n_nodes   = in_sizes[0] / HID;
  const int n_triples = in_sizes[2];

  // workspace layout
  char* ws = (char*)d_ws;
  float* agg = (float*)ws;                       size_t off = (size_t)n_nodes * HID * 4;
  float* nc  = (float*)(ws + off);               off += (size_t)n_nodes * 4;
  off = (off + 255) & ~(size_t)255;
  _Float16* W3h = (_Float16*)(ws + off);         off += HID * XK * 2;
  _Float16* Wuh = (_Float16*)(ws + off);         off += HID * CK * 2;

  // 1) zero agg + nc (nc directly follows agg)
  long zn = (long)n_nodes * (HID + 1);
  zero_kernel<<<1024, 256, 0, stream>>>(agg, zn);
  // 2) convert weights to f16 (padded layouts)
  prep_kernel<<<24, 256, 0, stream>>>(W3, Wu, W3h, Wuh);
  // 3) main WMMA triple kernel
  const int ntiles = (n_triples + TILE - 1) / TILE;
  const int blocks = 1250;
  const int wavesTotal = blocks * WAVES;
  const int iters = (ntiles + wavesTotal - 1) / wavesTotal;
  triples_kernel<<<blocks, WAVES * 32, 0, stream>>>(
      h, tb_index, tb_d1, tb_d2, W3h, Wuh, agg, nc,
      n_triples, ntiles, iters, wavesTotal);
  // 4) finalize: scale, residual, layernorm
  finalize_kernel<<<(n_nodes + 127) / 128, 128, 0, stream>>>(
      h, agg, nc, gamma, beta, out, n_nodes);
}
